// MultiHeadLatentAttention_15882789060803
// MI455X (gfx1250) — matrix operand order repaired and checked by reading, not compiled, not run on hardware
//
#include <hip/hip_runtime.h>

typedef __attribute__((ext_vector_type(16))) _Float16 v16h;
typedef __attribute__((ext_vector_type(8)))  _Float16 v8h;
typedef __attribute__((ext_vector_type(8)))  float    v8f;

constexpr int Bc  = 2;
constexpr int Sc  = 2048;
constexpr int HIDc = 2048;
constexpr int NHc = 16;
constexpr int HDc = 128;
constexpr int RDc = 64;
constexpr int Mrows = Bc * Sc;          // 4096
constexpr int QKD = HDc + RDc;          // 192

__device__ __forceinline__ v16h cat16(v8h lo, v8h hi) {
  return __builtin_shufflevector(lo, hi, 0,1,2,3,4,5,6,7,8,9,10,11,12,13,14,15);
}
__device__ __forceinline__ v8f wmma_f16(v16h a, v16h b, v8f c) {
  // D = A(16x32 f16) * B(32x16 f16) + C(16x16 f32)
  return __builtin_amdgcn_wmma_f32_16x16x32_f16(false, a, false, b, (short)0, c, false, false);
}

// Async global->LDS 16-byte copy (CDNA5, tracked on ASYNCcnt). Per-lane:
// copies 16 bytes from g to LDS offset lds_off. Inline asm keeps it portable across toolchains.
__device__ __forceinline__ void async_ld_lds_b128(const _Float16* g, unsigned lds_off) {
  asm volatile("global_load_async_to_lds_b128 %0, %1, off"
               :: "v"(lds_off), "v"(g)
               : "memory");
}
__device__ __forceinline__ void wait_asynccnt0() {
#if __has_builtin(__builtin_amdgcn_s_wait_asynccnt)
  __builtin_amdgcn_s_wait_asynccnt(0);
#else
  asm volatile("s_wait_asynccnt 0" ::: "memory");
#endif
}

// ---------------------------------------------------------------------------
// Tiled GEMM: C[M,N] = A[M,K] @ W[K,N] + bias[N]   (fp32 in/out, f16 WMMA core)
// Block: 256 threads (8 waves). Tile 128x128, K-step 32.
// Waves: 4 (M) x 2 (N); each wave computes 32x64 = 2x4 WMMA tiles.
// ---------------------------------------------------------------------------
__global__ void __launch_bounds__(256)
gemm_bias_f16(const float* __restrict__ A, const float* __restrict__ W,
              const float* __restrict__ bias, float* __restrict__ C,
              int M, int N, int K)
{
  __shared__ __align__(16) _Float16 lA[128][40];   // [m][k] row-major, pad 8
  __shared__ __align__(16) _Float16 lB[128][40];   // [n][k] (W transposed), pad 8

  const int t    = threadIdx.x;
  const int bm   = blockIdx.y * 128;
  const int bn   = blockIdx.x * 128;
  const int wave = t >> 5, lane = t & 31, g = lane >> 4, l16 = lane & 15;
  const int wm   = (wave & 3) * 32;
  const int wn   = (wave >> 2) * 64;

  v8f acc[2][4];
  #pragma unroll
  for (int mi = 0; mi < 2; ++mi)
    #pragma unroll
    for (int ni = 0; ni < 4; ++ni) {
      v8f z = {0.f,0.f,0.f,0.f,0.f,0.f,0.f,0.f};
      acc[mi][ni] = z;
    }

  const int arow = t >> 1, acs = (t & 1) * 16;       // A: 2 thr/row, 16 fp32 each
  const int bkr  = t & 31, bns = (t >> 5) * 16;      // W: 1 k-row seg, 16 fp32 each

  for (int kk = 0; kk < K; kk += 32) {
    { // stage A tile (convert fp32 -> f16)
      const float* src = A + (size_t)(bm + arow) * K + kk + acs;
      #pragma unroll
      for (int j = 0; j < 4; ++j) {
        float4 f = reinterpret_cast<const float4*>(src)[j];
        lA[arow][acs + j*4 + 0] = (_Float16)f.x;
        lA[arow][acs + j*4 + 1] = (_Float16)f.y;
        lA[arow][acs + j*4 + 2] = (_Float16)f.z;
        lA[arow][acs + j*4 + 3] = (_Float16)f.w;
      }
      if (kk + 32 < K) __builtin_prefetch(src + 32, 0, 0);   // global_prefetch_b8
    }
    { // stage W tile transposed: lB[n][k]
      const float* src = W + (size_t)(kk + bkr) * N + bn + bns;
      #pragma unroll
      for (int j = 0; j < 4; ++j) {
        float4 f = reinterpret_cast<const float4*>(src)[j];
        lB[bns + j*4 + 0][bkr] = (_Float16)f.x;
        lB[bns + j*4 + 1][bkr] = (_Float16)f.y;
        lB[bns + j*4 + 2][bkr] = (_Float16)f.z;
        lB[bns + j*4 + 3][bkr] = (_Float16)f.w;
      }
      if (kk + 32 < K)
        __builtin_prefetch(W + (size_t)(kk + 32 + bkr) * N + bn + bns, 0, 0);
    }
    __syncthreads();

    // A fragment: lane holds row M=l16; K split {g*8..+7, 16+g*8..+7} (ISA 7.12.2)
    v16h af[2], bf[4];
    #pragma unroll
    for (int mi = 0; mi < 2; ++mi) {
      const int r = wm + mi*16 + l16;
      v8h lo = *reinterpret_cast<const v8h*>(&lA[r][g*8]);
      v8h hi = *reinterpret_cast<const v8h*>(&lA[r][16 + g*8]);
      af[mi] = cat16(lo, hi);
    }
    // B fragment: lane holds col N=l16; K split {g*8..+7, 16+g*8..+7}
    #pragma unroll
    for (int ni = 0; ni < 4; ++ni) {
      const int c = wn + ni*16 + l16;
      v8h lo = *reinterpret_cast<const v8h*>(&lB[c][g*8]);
      v8h hi = *reinterpret_cast<const v8h*>(&lB[c][16 + g*8]);
      bf[ni] = cat16(lo, hi);
    }
    #pragma unroll
    for (int mi = 0; mi < 2; ++mi)
      #pragma unroll
      for (int ni = 0; ni < 4; ++ni)
        acc[mi][ni] = wmma_f16(af[mi], bf[ni], acc[mi][ni]);
    __syncthreads();
  }

  // C/D layout: lane => col = l16, VGPR v => row = v + g*8
  #pragma unroll
  for (int ni = 0; ni < 4; ++ni) {
    const int col = bn + wn + ni*16 + l16;
    const float bv = bias[col];
    #pragma unroll
    for (int mi = 0; mi < 2; ++mi)
      #pragma unroll
      for (int v = 0; v < 8; ++v) {
        const int row = bm + wm + mi*16 + g*8 + v;
        C[(size_t)row * N + col] = acc[mi][ni][v] + bv;
      }
  }
}

// ---------------------------------------------------------------------------
// Pack: raw fp32 [B*S, NH*128] (b,s,h,d) -> f16 [B,NH,S,dstW] at column dOff
// ---------------------------------------------------------------------------
__global__ void pack_head_f16(const float* __restrict__ raw,
                              _Float16* __restrict__ dst, int dstW, int dOff)
{
  size_t i = (size_t)blockIdx.x * blockDim.x + threadIdx.x;
  if (i >= (size_t)Mrows * NHc * HDc) return;
  const int d  = (int)(i & (HDc - 1));
  size_t r     = i >> 7;
  const int h  = (int)(r & (NHc - 1));
  const size_t bs = r >> 4;
  const int s  = (int)(bs % Sc);
  const int b  = (int)(bs / Sc);
  dst[(((size_t)b * NHc + h) * Sc + s) * dstW + dOff + d] =
      (_Float16)raw[bs * (size_t)(NHc * HDc) + (size_t)h * HDc + d];
}

// ---------------------------------------------------------------------------
// RoPE pack: raw fp32 [B*S, NH*64] -> f16 [B,NH,S,192] at columns 128..191
// ---------------------------------------------------------------------------
__global__ void pack_rope_f16(const float* __restrict__ raw,
                              _Float16* __restrict__ dst)
{
  size_t i = (size_t)blockIdx.x * blockDim.x + threadIdx.x;
  if (i >= (size_t)Mrows * NHc * (RDc / 2)) return;
  const int j  = (int)(i & 31);
  size_t r     = i >> 5;
  const int h  = (int)(r & (NHc - 1));
  const size_t bs = r >> 4;
  const int s  = (int)(bs % Sc);
  const int b  = (int)(bs / Sc);

  const float* src = raw + bs * (size_t)(NHc * RDc) + (size_t)h * RDc;
  const float t1 = src[j];
  const float t2 = src[j + RDc / 2];
  // inv_freq = 10000^(-2j/RD)
  const float invf = __expf(-((float)(2 * j) / (float)RDc) * 9.210340371976184f);
  const float ang = (float)s * invf;
  float sn, cs;
  __sincosf(ang, &sn, &cs);
  _Float16* drow = dst + (((size_t)b * NHc + h) * Sc + s) * QKD + HDc;
  drow[j]            = (_Float16)(t1 * cs - t2 * sn);
  drow[j + RDc / 2]  = (_Float16)(t2 * cs + t1 * sn);
}

// ---------------------------------------------------------------------------
// Flash attention (causal). Block = 128 threads = 4 waves; 64 query rows/block
// (16 rows per wave). Key loop in 32-key steps: K tile staged with
// GLOBAL_LOAD_ASYNC_TO_LDS_B128 (ASYNCcnt), V tile transposed through VGPRs.
// QK^T (12 WMMA) -> online softmax (16-lane shfl_xor) -> P via per-wave LDS
// -> PV (8 WMMA).
// ---------------------------------------------------------------------------
__global__ void __launch_bounds__(128)
mla_attn(const _Float16* __restrict__ Q, const _Float16* __restrict__ K,
         const _Float16* __restrict__ V, float* __restrict__ ctx)
{
  __shared__ __align__(16) _Float16 lK[32][200];      // [key][d], pad 8
  __shared__ __align__(16) _Float16 lV[128][40];      // [n][key] transposed, pad 8
  __shared__ __align__(16) _Float16 lP[4][16][40];    // per-wave P tile [qrow][key]

  const int t = threadIdx.x;
  const int wave = t >> 5, lane = t & 31, g = lane >> 4, l16 = lane & 15;
  const int h = blockIdx.y, b = blockIdx.z;
  const int bh = b * NHc + h;
  const int qbase = blockIdx.x * 64 + wave * 16;
  const float scale = 0.07216878364870322f;           // 1/sqrt(192)

  const _Float16* qp = Q + (size_t)bh * Sc * QKD;
  const _Float16* kp = K + (size_t)bh * Sc * QKD;
  const _Float16* vp = V + (size_t)bh * Sc * HDc;

  // Q fragments (held in registers whole kernel): 6 slices of K-dim 32
  v16h qf[6];
  {
    const _Float16* qrow = qp + (size_t)(qbase + l16) * QKD;
    #pragma unroll
    for (int s6 = 0; s6 < 6; ++s6) {
      v8h lo = *reinterpret_cast<const v8h*>(qrow + s6*32 + g*8);
      v8h hi = *reinterpret_cast<const v8h*>(qrow + s6*32 + 16 + g*8);
      qf[s6] = cat16(lo, hi);
    }
  }

  v8f acc[8];
  #pragma unroll
  for (int n = 0; n < 8; ++n) {
    v8f z = {0.f,0.f,0.f,0.f,0.f,0.f,0.f,0.f};
    acc[n] = z;
  }
  float mrow[8], lrow[8];
  #pragma unroll
  for (int v = 0; v < 8; ++v) { mrow[v] = -1e30f; lrow[v] = 0.f; }

  const int kend  = blockIdx.x * 64 + 64;   // causal bound for this block
  const int lkey  = t >> 2;                 // 0..31: key row this thread stages
  const int ldseg = (t & 3) * 48;           // K tile: 48 halves per thread
  const int lnseg = (t & 3) * 32;           // V tile: 32 halves per thread

  for (int kb = 0; kb < kend; kb += 32) {
    { // async-stage K tile [32][192]: pure f16 copy, no VGPR round-trip
      const _Float16* src = kp + (size_t)(kb + lkey) * QKD + ldseg;
      #pragma unroll
      for (int j = 0; j < 6; ++j)
        async_ld_lds_b128(src + j*8,
                          (unsigned)(size_t)&lK[lkey][ldseg + j*8]);
    }
    { // stage V tile transposed: lV[n][key] (needs rearrangement -> VGPR path)
      const _Float16* src = vp + (size_t)(kb + lkey) * HDc + lnseg;
      #pragma unroll
      for (int j = 0; j < 4; ++j) {
        v8h x = *reinterpret_cast<const v8h*>(src + j*8);
        #pragma unroll
        for (int e = 0; e < 8; ++e) lV[lnseg + j*8 + e][lkey] = x[e];
      }
    }
    wait_asynccnt0();          // this wave's async LDS writes have landed
    __syncthreads();           // all waves' writes visible

    // scores: two 16-key tiles
    float val0[8], val1[8];
    #pragma unroll
    for (int kt = 0; kt < 2; ++kt) {
      v8f sc = {0.f,0.f,0.f,0.f,0.f,0.f,0.f,0.f};
      const int keyl = kt*16 + l16;
      #pragma unroll
      for (int s6 = 0; s6 < 6; ++s6) {
        v8h lo = *reinterpret_cast<const v8h*>(&lK[keyl][s6*32 + g*8]);
        v8h hi = *reinterpret_cast<const v8h*>(&lK[keyl][s6*32 + 16 + g*8]);
        sc = wmma_f16(qf[s6], cat16(lo, hi), sc);
      }
      const int kcol = kb + keyl;
      float* dv = kt ? val1 : val0;
      #pragma unroll
      for (int v = 0; v < 8; ++v) {
        const int row = qbase + g*8 + v;
        dv[v] = (kcol <= row) ? sc[v] * scale : -1e30f;
      }
    }

    // online softmax per row (row stats replicated across the 16 lanes of group g)
    #pragma unroll
    for (int v = 0; v < 8; ++v) {
      float rmax = fmaxf(val0[v], val1[v]);
      #pragma unroll
      for (int msk = 1; msk < 16; msk <<= 1)
        rmax = fmaxf(rmax, __shfl_xor(rmax, msk, 32));
      const float nm   = fmaxf(mrow[v], rmax);
      const float corr = __expf(mrow[v] - nm);
      mrow[v] = nm;
      const float p0 = __expf(val0[v] - nm);
      const float p1 = __expf(val1[v] - nm);
      float rs = p0 + p1;
      #pragma unroll
      for (int msk = 1; msk < 16; msk <<= 1)
        rs += __shfl_xor(rs, msk, 32);
      lrow[v] = lrow[v] * corr + rs;
      #pragma unroll
      for (int n = 0; n < 8; ++n) acc[n][v] *= corr;
      lP[wave][g*8 + v][l16]      = (_Float16)p0;
      lP[wave][g*8 + v][16 + l16] = (_Float16)p1;
    }

    // P (16x32) @ V (32x128): re-load P in A-operand layout from LDS
    v16h pf;
    {
      v8h lo = *reinterpret_cast<const v8h*>(&lP[wave][l16][g*8]);
      v8h hi = *reinterpret_cast<const v8h*>(&lP[wave][l16][16 + g*8]);
      pf = cat16(lo, hi);
    }
    #pragma unroll
    for (int n = 0; n < 8; ++n) {
      v8h lo = *reinterpret_cast<const v8h*>(&lV[n*16 + l16][g*8]);
      v8h hi = *reinterpret_cast<const v8h*>(&lV[n*16 + l16][16 + g*8]);
      acc[n] = wmma_f16(pf, cat16(lo, hi), acc[n]);
    }
    __syncthreads();   // all reads done before next iteration's staging
  }

  // write ctx as [B*S, NH*HD] (already in (b,s,h,d) order for the output GEMM)
  #pragma unroll
  for (int n = 0; n < 8; ++n)
    #pragma unroll
    for (int v = 0; v < 8; ++v) {
      const int row = qbase + g*8 + v;
      ctx[((size_t)b * Sc + row) * (size_t)(NHc * HDc) + h * HDc + n*16 + l16] =
          acc[n][v] / lrow[v];
    }
}

// ---------------------------------------------------------------------------
// Host launcher
// ---------------------------------------------------------------------------
extern "C" void kernel_launch(void* const* d_in, const int* in_sizes, int n_in,
                              void* d_out, int out_size, void* d_ws, size_t ws_size,
                              hipStream_t stream)
{
  (void)in_sizes; (void)n_in; (void)out_size; (void)ws_size;
  const float* x     = (const float*)d_in[0];
  const float* W_kvd = (const float*)d_in[1];
  const float* b_kvd = (const float*)d_in[2];
  const float* W_ku  = (const float*)d_in[3];
  const float* b_ku  = (const float*)d_in[4];
  const float* W_vu  = (const float*)d_in[5];
  const float* b_vu  = (const float*)d_in[6];
  const float* W_kr  = (const float*)d_in[7];
  const float* b_kr  = (const float*)d_in[8];
  const float* W_qd  = (const float*)d_in[9];
  const float* b_qd  = (const float*)d_in[10];
  const float* W_qu  = (const float*)d_in[11];
  const float* b_qu  = (const float*)d_in[12];
  const float* W_qr  = (const float*)d_in[13];
  const float* b_qr  = (const float*)d_in[14];
  const float* W_o   = (const float*)d_in[15];
  const float* b_o   = (const float*)d_in[16];

  char* ws = (char*)d_ws;
  constexpr size_t OFF_KV  = 0;
  constexpr size_t OFF_QC  = OFF_KV  + (size_t)Mrows * 512  * 4;
  constexpr size_t OFF_RAW = OFF_QC  + (size_t)Mrows * 1536 * 4;
  constexpr size_t OFF_Q16 = OFF_RAW + (size_t)Mrows * 2048 * 4;
  constexpr size_t OFF_K16 = OFF_Q16 + (size_t)Bc*NHc*Sc*QKD * 2;
  constexpr size_t OFF_V16 = OFF_K16 + (size_t)Bc*NHc*Sc*QKD * 2;

  float*    kv  = (float*)(ws + OFF_KV);
  float*    qc  = (float*)(ws + OFF_QC);
  float*    raw = (float*)(ws + OFF_RAW);
  _Float16* q16 = (_Float16*)(ws + OFF_Q16);
  _Float16* k16 = (_Float16*)(ws + OFF_K16);
  _Float16* v16 = (_Float16*)(ws + OFF_V16);
  float*    ctx = raw;  // raw region is free by attention time

  auto gemm = [&](const float* A, const float* W, const float* bb, float* C,
                  int M, int N, int Kd) {
    dim3 grid(N / 128, M / 128);
    gemm_bias_f16<<<grid, 256, 0, stream>>>(A, W, bb, C, M, N, Kd);
  };
  const int packBlocks = (int)(((size_t)Mrows * NHc * HDc + 255) / 256);
  const int ropeBlocks = (int)(((size_t)Mrows * NHc * (RDc/2) + 255) / 256);

  // projections
  gemm(x,  W_kvd, b_kvd, kv,  Mrows, 512,  HIDc);            // kv_c
  gemm(x,  W_qd,  b_qd,  qc,  Mrows, 1536, HIDc);            // q_c
  gemm(kv, W_ku,  b_ku,  raw, Mrows, 2048, 512);             // keys_c
  pack_head_f16<<<packBlocks, 256, 0, stream>>>(raw, k16, QKD, 0);
  gemm(kv, W_vu,  b_vu,  raw, Mrows, 2048, 512);             // values
  pack_head_f16<<<packBlocks, 256, 0, stream>>>(raw, v16, HDc, 0);
  gemm(kv, W_kr,  b_kr,  raw, Mrows, 1024, 512);             // keys_r
  pack_rope_f16<<<ropeBlocks, 256, 0, stream>>>(raw, k16);
  gemm(qc, W_qu,  b_qu,  raw, Mrows, 2048, 1536);            // queries_c
  pack_head_f16<<<packBlocks, 256, 0, stream>>>(raw, q16, QKD, 0);
  gemm(qc, W_qr,  b_qr,  raw, Mrows, 1024, 1536);            // queries_r
  pack_rope_f16<<<ropeBlocks, 256, 0, stream>>>(raw, q16);

  // attention
  mla_attn<<<dim3(Sc / 64, NHc, Bc), 128, 0, stream>>>(q16, k16, v16, ctx);

  // output projection
  gemm(ctx, W_o, b_o, (float*)d_out, Mrows, HIDc, NHc * HDc);
}
